// Attention_27358941675632
// MI455X (gfx1250) — matrix operand order repaired and checked by reading, not compiled, not run on hardware
//
#include <hip/hip_runtime.h>

#define B_SZ  32
#define T_SZ  4096
#define CTXW  1024
#define DECH  512
#define ATTH  256

typedef __attribute__((ext_vector_type(16))) __bf16 v16bf;
typedef __attribute__((ext_vector_type(8)))  float  v8f;

union FragBF {
    v16bf          v;
    unsigned short u[16];
    uint4          q[2];
};

__device__ __forceinline__ unsigned short f2bf(float f) {
    unsigned u = __float_as_uint(f);
    u += 0x7FFFu + ((u >> 16) & 1u);     // round-to-nearest-even
    return (unsigned short)(u >> 16);
}

// Async global -> LDS (CDNA5): 16B per lane, tracked by ASYNCcnt.
__device__ __forceinline__ void async_load_b128(unsigned lds_off, const void* gptr) {
    asm volatile("global_load_async_to_lds_b128 %0, %1, off"
                 :: "v"(lds_off), "v"(gptr) : "memory");
}

// --------------------------------------------------------------------------
// Kernel 0: one-time U_w f32 -> bf16 conversion into workspace.
// --------------------------------------------------------------------------
__global__ __launch_bounds__(256) void cvt_uw_kernel(
    const float* __restrict__ U_w, unsigned short* __restrict__ u_bf) {
    const size_t i = (size_t)blockIdx.x * blockDim.x + threadIdx.x;  // float4 index
    const float4 f = ((const float4*)U_w)[i];
    uint2 pk;
    pk.x = (unsigned)f2bf(f.x) | ((unsigned)f2bf(f.y) << 16);
    pk.y = (unsigned)f2bf(f.z) | ((unsigned)f2bf(f.w) << 16);
    ((uint2*)u_bf)[i] = pk;
}

// --------------------------------------------------------------------------
// Kernel 1: s1[b,h] = prev[b,:] . W_w[h,:] + W_b[h] + U_b[h]
// --------------------------------------------------------------------------
__global__ __launch_bounds__(ATTH) void tmp1_kernel(
    const float* __restrict__ prev, const float* __restrict__ W_w,
    const float* __restrict__ W_b,  const float* __restrict__ U_b,
    float* __restrict__ s1) {
    __shared__ float sprev[DECH];
    const int b = blockIdx.x;
    const int h = threadIdx.x;
    for (int k = threadIdx.x; k < DECH; k += blockDim.x)
        sprev[k] = prev[(size_t)b * DECH + k];
    __syncthreads();
    float acc = 0.f;
    const float4* wr = (const float4*)(W_w + (size_t)h * DECH);
    #pragma unroll 4
    for (int k = 0; k < DECH / 4; ++k) {
        float4 w = wr[k];
        acc += w.x * sprev[4 * k] + w.y * sprev[4 * k + 1] +
               w.z * sprev[4 * k + 2] + w.w * sprev[4 * k + 3];
    }
    s1[(size_t)b * ATTH + h] = acc + W_b[h] + U_b[h];
}

// --------------------------------------------------------------------------
// Kernel 2: energy[b,t] = sum_h relu(enc[b,t,:].U_w[h,:] + s1[b,h]) * v[h]
// Workgroup = 64 t-rows x all 256 h for one batch; 16 waves, one 16-h slab
// each. Double-buffered async global->LDS staging of enc (f32), LDS-side
// bf16 convert, then bf16 WMMA 16x16x32 with fp32 accumulate over K=1024.
// --------------------------------------------------------------------------
__global__ __launch_bounds__(512) void energy_kernel(
    const float* __restrict__ enc, const unsigned short* __restrict__ u_bf,
    const float* __restrict__ s1,  const float* __restrict__ v,
    float* __restrict__ energy) {
    __shared__ float          fbuf[2][64][64];  // double-buffered f32 staging (2 x 16KB)
    __shared__ unsigned short tile[64][80];     // bf16 WMMA tile; 160B row stride (16B aligned)
    __shared__ float          lds_part[16][64]; // per-wave partials (deterministic reduce)

    const int b    = blockIdx.y;
    const int t0   = blockIdx.x * 64;
    const int tid  = threadIdx.x;
    const int wave = tid >> 5;
    const int lane = tid & 31;
    const int lm   = lane & 15;               // N / row-within-16
    const int lhi  = lane >> 4;               // K-half select
    const int kb   = lhi * 8;

    const int   h   = (wave << 4) + lm;       // this lane's h column
    const float s1h = s1[(size_t)b * ATTH + h];
    const float vh  = v[h];

    v8f acc[4] = {};                          // 4 t-subtiles x (16x16) fp32

    const size_t encBase = ((size_t)b * T_SZ + t0) * CTXW;
    const int lrow = tid >> 3;                // 0..63 : staging row
    const int lcol = (tid & 7) * 8;           // 0..56 : staging col (8 floats)
    const float* encRowPtr = enc + encBase + (size_t)lrow * CTXW + lcol;

    const unsigned ldsOff0 = (unsigned)(uintptr_t)&fbuf[0][lrow][lcol];
    const unsigned ldsOff1 = (unsigned)(uintptr_t)&fbuf[1][lrow][lcol];

    // ---- prologue: async-stage chunk 0 ----
    async_load_b128(ldsOff0,      encRowPtr);
    async_load_b128(ldsOff0 + 16, encRowPtr + 4);

    for (int kc = 0; kc < CTXW; kc += 64) {
        const int      cur     = (kc >> 6) & 1;
        const unsigned nextOff = cur ? ldsOff0 : ldsOff1;

        if (kc + 64 < CTXW) {
            // issue next chunk's async loads, then wait for current's (in-order: <=2 left)
            const float* g = encRowPtr + kc + 64;
            async_load_b128(nextOff,      g);
            async_load_b128(nextOff + 16, g + 4);
            asm volatile("s_wait_asynccnt 0x2" ::: "memory");
        } else {
            asm volatile("s_wait_asynccnt 0x0" ::: "memory");
        }
        __syncthreads();   // all waves staged current chunk; prev compute done

        // ---- convert current f32 chunk -> bf16 WMMA tile ----
        const float4 f0 = *(const float4*)&fbuf[cur][lrow][lcol];
        const float4 f1 = *(const float4*)&fbuf[cur][lrow][lcol + 4];
        uint4 pk;
        pk.x = (unsigned)f2bf(f0.x) | ((unsigned)f2bf(f0.y) << 16);
        pk.y = (unsigned)f2bf(f0.z) | ((unsigned)f2bf(f0.w) << 16);
        pk.z = (unsigned)f2bf(f1.x) | ((unsigned)f2bf(f1.y) << 16);
        pk.w = (unsigned)f2bf(f1.z) | ((unsigned)f2bf(f1.w) << 16);
        *(uint4*)&tile[lrow][lcol] = pk;
        __syncthreads();

        // ---- 2 K-steps of 32; 4 t-subtiles each ----
        #pragma unroll
        for (int ks = 0; ks < 64; ks += 32) {
            // B fragment: pre-converted bf16, N = h, K = kc+ks+kb .. +8 and kc+ks+16+kb .. +8
            FragBF bf;
            const uint4* ub = (const uint4*)(u_bf + (size_t)h * CTXW + kc + ks + kb);
            bf.q[0] = ub[0];
            bf.q[1] = ub[2];
            // A fragments for all 4 t-subtiles
            FragBF af[4];
            #pragma unroll
            for (int s = 0; s < 4; ++s) {
                const int r = s * 16 + lm;
                af[s].q[0] = *(const uint4*)&tile[r][ks + kb];
                af[s].q[1] = *(const uint4*)&tile[r][ks + 16 + kb];
            }
            #pragma unroll
            for (int s = 0; s < 4; ++s)
                acc[s] = __builtin_amdgcn_wmma_f32_16x16x32_bf16(
                    false, af[s].v, false, bf.v, (short)0, acc[s], false, false);

            // Force issue shape: B-frag VMEM reads, then ALL A-frag DS reads,
            // then the 4 WMMAs back-to-back (partial dscnt waits, loads overlap
            // the matrix pipe instead of a full s_wait_dscnt 0 per WMMA).
            __builtin_amdgcn_sched_group_barrier(0x020, 2, 0);  // 2 VMEM reads
            __builtin_amdgcn_sched_group_barrier(0x100, 8, 0);  // 8 DS reads
            __builtin_amdgcn_sched_group_barrier(0x008, 4, 0);  // 4 WMMA
        }
    }

    // ---- epilogue: relu(+s1)*v, reduce the 16 h of each lane-half ----
    #pragma unroll
    for (int s = 0; s < 4; ++s) {
        #pragma unroll
        for (int r = 0; r < 8; ++r) {
            float e = fmaxf(acc[s][r] + s1h, 0.f) * vh;
            e += __shfl_xor(e, 1, 32);
            e += __shfl_xor(e, 2, 32);
            e += __shfl_xor(e, 4, 32);
            e += __shfl_xor(e, 8, 32);        // each 16-lane half holds its h-sum
            if (lm == 0)
                lds_part[wave][s * 16 + r + (lhi << 3)] = e;
        }
    }
    __syncthreads();
    if (tid < 64) {
        float e = 0.f;
        #pragma unroll
        for (int w = 0; w < 16; ++w) e += lds_part[w][tid];
        energy[(size_t)b * T_SZ + t0 + tid] = e;
    }
}

// --------------------------------------------------------------------------
// Kernel 3: in-place softmax over T for each batch row
// --------------------------------------------------------------------------
__global__ __launch_bounds__(256) void softmax_kernel(float* __restrict__ e) {
    __shared__ float red[256];
    const int b = blockIdx.x, tid = threadIdx.x;
    float* row = e + (size_t)b * T_SZ;

    float m = -1e30f;
    for (int t = tid; t < T_SZ; t += 256) m = fmaxf(m, row[t]);
    red[tid] = m;
    __syncthreads();
    for (int s = 128; s > 0; s >>= 1) {
        if (tid < s) red[tid] = fmaxf(red[tid], red[tid + s]);
        __syncthreads();
    }
    m = red[0];
    __syncthreads();

    float sum = 0.f;
    for (int t = tid; t < T_SZ; t += 256) {
        float x = __expf(row[t] - m);
        row[t] = x;
        sum += x;
    }
    red[tid] = sum;
    __syncthreads();
    for (int s = 128; s > 0; s >>= 1) {
        if (tid < s) red[tid] += red[tid + s];
        __syncthreads();
    }
    const float inv = 1.f / red[0];
    for (int t = tid; t < T_SZ; t += 256) row[t] *= inv;
}

// --------------------------------------------------------------------------
// Kernel 4: c[b,c] = sum_t alpha[b,t] * enc[b,t,c]   (bandwidth pass)
// --------------------------------------------------------------------------
__global__ __launch_bounds__(256) void context_kernel(
    const float* __restrict__ enc, const float* __restrict__ alpha,
    float* __restrict__ out) {
    __shared__ float sa[256];
    const int b = blockIdx.x, tid = threadIdx.x;
    const int c4 = tid * 4;
    float4 acc = {0.f, 0.f, 0.f, 0.f};
    for (int tb = 0; tb < T_SZ; tb += 256) {
        __syncthreads();
        sa[tid] = alpha[(size_t)b * T_SZ + tb + tid];
        __syncthreads();
        const float4* ep = (const float4*)(enc + ((size_t)b * T_SZ + tb) * CTXW + c4);
        for (int j = 0; j < 256; ++j) {
            const float a  = sa[j];
            const float4 e = ep[(size_t)j * (CTXW / 4)];
            acc.x += a * e.x; acc.y += a * e.y;
            acc.z += a * e.z; acc.w += a * e.w;
        }
    }
    *(float4*)(out + (size_t)b * CTXW + c4) = acc;
}

// --------------------------------------------------------------------------
extern "C" void kernel_launch(void* const* d_in, const int* in_sizes, int n_in,
                              void* d_out, int out_size, void* d_ws, size_t ws_size,
                              hipStream_t stream) {
    // setup_inputs order: batch_size, max_seq_len, prev, enc, W_w, W_b, U_w, U_b, v
    const float* prev = (const float*)d_in[2];
    const float* enc  = (const float*)d_in[3];
    const float* W_w  = (const float*)d_in[4];
    const float* W_b  = (const float*)d_in[5];
    const float* U_w  = (const float*)d_in[6];
    const float* U_b  = (const float*)d_in[7];
    const float* vv   = (const float*)d_in[8];
    float* out = (float*)d_out;

    // workspace layout (all 16B-aligned):
    //   s1     : B*ATTH floats            (32 KB)
    //   u_bf   : ATTH*CTXW bf16           (512 KB)
    //   energy : B*T floats, softmax in place -> alpha (512 KB)
    float*          s1     = (float*)d_ws;
    unsigned short* u_bf   = (unsigned short*)(s1 + B_SZ * ATTH);
    float*          energy = (float*)(u_bf + (size_t)ATTH * CTXW);

    cvt_uw_kernel<<<dim3((ATTH * CTXW / 4) / 256), dim3(256), 0, stream>>>(U_w, u_bf);
    tmp1_kernel<<<dim3(B_SZ), dim3(ATTH), 0, stream>>>(prev, W_w, W_b, U_b, s1);
    energy_kernel<<<dim3(T_SZ / 64, B_SZ), dim3(512), 0, stream>>>(enc, u_bf, s1, vv, energy);
    softmax_kernel<<<dim3(B_SZ), dim3(256), 0, stream>>>(energy);
    context_kernel<<<dim3(B_SZ), dim3(256), 0, stream>>>(enc, energy, out);
}
